// Rgs_Branch_89077621719623
// MI455X (gfx1250) — matrix operand order repaired and checked by reading, not compiled, not run on hardware
//
#include <hip/hip_runtime.h>

#define DIM   256
#define TOPK  16
#define BATCH 16
#define TLEN  8192
#define NPAT  4096

typedef __attribute__((ext_vector_type(16))) __bf16 v16bf;
typedef __attribute__((ext_vector_type(8)))  float  v8f;

struct __align__(16) u32x4 { unsigned x, y, z, w; };

union BFPack {
  unsigned short u[16];
  u32x4          q[2];
  v16bf          v;
};

// Tensor Data Mover path: 5-arg builtin verified on ROCm 7.2 (clang-22).
// clang-23 (amdgpu-toolchain) uses a 6-arg form -> fall back to plain copy there.
#if defined(__has_builtin)
#if __has_builtin(__builtin_amdgcn_tensor_load_to_lds) && \
    __has_builtin(__builtin_amdgcn_s_wait_tensorcnt) && (__clang_major__ < 23)
#define USE_TDM 1
#endif
#endif
#ifndef USE_TDM
#define USE_TDM 0
#endif

#if USE_TDM
typedef __attribute__((ext_vector_type(4))) unsigned int tdm_u32x4;
typedef __attribute__((ext_vector_type(8))) int          tdm_i32x8;
typedef __attribute__((ext_vector_type(4))) int          tdm_i32x4;
#endif

__device__ __forceinline__ unsigned short f2bf(float f) {
  unsigned u = __float_as_uint(f);
  u += 0x7FFFu + ((u >> 16) & 1u);          // round-to-nearest-even
  return (unsigned short)(u >> 16);
}

__device__ __forceinline__ float gelu_exact(float x) {
  return 0.5f * x * (1.0f + erff(x * 0.70710678118654752440f));
}

// ---- WMMA helpers (CDNA5 16x16x32 bf16, f32 accum) -------------------------
__device__ __forceinline__ v8f wmma_bf16(v16bf a, v16bf b, v8f c) {
  // 8 args: (neg_a, A, neg_b, B, c_mod, C, reuse_a, reuse_b)
  return __builtin_amdgcn_wmma_f32_16x16x32_bf16(false, a, false, b, (short)0, c,
                                                 false, false);
}

// A-matrix 16x32 bf16: lane<16 holds K k0..k0+15, lane>=16 holds
// k0+16..k0+31; row = lane&15 (matches B k order).
__device__ __forceinline__ v16bf loadA_u16(const unsigned short* src, int ld, int k0) {
  int lane = threadIdx.x & 31;
  const unsigned short* p = src + (lane & 15) * ld + k0 + ((lane >> 4) << 4);
  BFPack t;
  t.q[0] = *(const u32x4*)p;
  t.q[1] = *(const u32x4*)(p + 8);
  return t.v;
}

// Same A layout from f32 source with on-the-fly bf16 conversion.
__device__ __forceinline__ v16bf loadA_f32cvt(const float* src, int ld, int k0) {
  int lane = threadIdx.x & 31;
  const float* p = src + (lane & 15) * ld + k0 + ((lane >> 4) << 4);
  BFPack t;
#pragma unroll
  for (int i = 0; i < 8; ++i) t.u[i]     = f2bf(p[i]);
#pragma unroll
  for (int i = 0; i < 8; ++i) t.u[8 + i] = f2bf(p[8 + i]);
  return t.v;
}

// B-matrix 32x16 bf16: column n = lane&15, lane<16 holds K k0..k0+15,
// lane>=16 holds K k0+16..k0+31. Element B[k][n] = buf[(n0+n)*stride + k].
__device__ __forceinline__ v16bf loadB_u16(const unsigned short* base, long stride,
                                           int n0, int k0) {
  int lane = threadIdx.x & 31;
  const unsigned short* p = base + (long)(n0 + (lane & 15)) * stride + k0 + ((lane >> 4) << 4);
  BFPack t;
  t.q[0] = *(const u32x4*)p;
  t.q[1] = *(const u32x4*)(p + 8);
  return t.v;
}

// B-matrix from f32 weight W[n][k] row-major with conversion (W1/W2 paths).
__device__ __forceinline__ v16bf loadB_f32cvt(const float* w, int ld, int n0, int k0) {
  int lane = threadIdx.x & 31;
  const float* p = w + (long)(n0 + (lane & 15)) * ld + k0 + ((lane >> 4) << 4);
  BFPack t;
#pragma unroll
  for (int i = 0; i < 16; ++i) t.u[i] = f2bf(p[i]);
  return t.v;
}

// Cross-lane reductions within 16-lane halves / full wave32.
__device__ __forceinline__ float redmax16(float v) {
#pragma unroll
  for (int m = 1; m < 16; m <<= 1) v = fmaxf(v, __shfl_xor(v, m, 32));
  return v;
}
__device__ __forceinline__ float redsum16(float v) {
#pragma unroll
  for (int m = 1; m < 16; m <<= 1) v += __shfl_xor(v, m, 32);
  return v;
}
__device__ __forceinline__ float redsum32(float v) {
#pragma unroll
  for (int m = 1; m < 32; m <<= 1) v += __shfl_xor(v, m, 32);
  return v;
}

// ---- Kernel 1: per-batch top-16 (set only; order irrelevant downstream) ----
__global__ void __launch_bounds__(256) topk_kernel(const float* __restrict__ a_hat,
                                                   int* __restrict__ idx_out) {
  __shared__ float sv[NPAT];
  __shared__ float srv[256];
  __shared__ int   sri[256];
  int b = blockIdx.x, tid = threadIdx.x;
  for (int i = tid; i < NPAT; i += 256) sv[i] = a_hat[(long)b * NPAT + i];
  __syncthreads();
  for (int r = 0; r < TOPK; ++r) {
    float bv = -__builtin_inff();
    int bi = 0;
    for (int i = tid; i < NPAT; i += 256) {
      float v = sv[i];
      if (v > bv) { bv = v; bi = i; }
    }
    srv[tid] = bv; sri[tid] = bi;
    __syncthreads();
    for (int s = 128; s > 0; s >>= 1) {
      if (tid < s && srv[tid + s] > srv[tid]) { srv[tid] = srv[tid + s]; sri[tid] = sri[tid + s]; }
      __syncthreads();
    }
    if (tid == 0) { idx_out[b * TOPK + r] = sri[0]; sv[sri[0]] = -__builtin_inff(); }
    __syncthreads();
  }
}

// ---- Kernel 2: gather pattern_subset + q = psub @ Wq.T (tiny) --------------
__global__ void __launch_bounds__(DIM) qproj_kernel(const float* __restrict__ patterns,
                                                    const float* __restrict__ Wq,
                                                    const int* __restrict__ idx,
                                                    float* __restrict__ psub,
                                                    unsigned short* __restrict__ qbf) {
  __shared__ float sp[DIM];
  int g = blockIdx.x, o = threadIdx.x;
  int pi = idx[g];
  float pv = patterns[(long)pi * DIM + o];
  sp[o] = pv;
  psub[(long)g * DIM + o] = pv;
  __syncthreads();
  float acc = 0.0f;
  const float* wr = Wq + (long)o * DIM;
#pragma unroll 4
  for (int d = 0; d < DIM; ++d) acc += sp[d] * wr[d];
  qbf[(long)g * DIM + o] = f2bf(acc);
}

// ---- Kernel 2b: one-time Wkv f32 -> bf16 (feeds TDM / LDS staging) ---------
__global__ void __launch_bounds__(256) wkvcvt_kernel(const float* __restrict__ Wkv,
                                                     unsigned short* __restrict__ wkvbf) {
  int i = blockIdx.x * 256 + threadIdx.x;
  wkvbf[i] = f2bf(Wkv[i]);
}

// ---- Kernel 3: KV projection (WMMA), K:[b][t][d] bf16, V transposed [b][d][t]
__global__ void __launch_bounds__(256) kv_kernel(const float* __restrict__ feature,
                                                 const unsigned short* __restrict__ wkvbf,
                                                 unsigned short* __restrict__ Kbuf,
                                                 unsigned short* __restrict__ Vbuf) {
  extern __shared__ unsigned short smW[];  // 512 x 256 bf16 = 256 KB (CDNA5: 320KB LDS/WGP)
  int tid = threadIdx.x;
  int wave = tid >> 5, lane = tid & 31, half = lane >> 4, col = lane & 15;

#if USE_TDM
  if (wave == 0) {
    // Tensor DMA: 2D tensor 512 rows x 256 cols of 2-byte elems, tile == tensor,
    // global -> LDS. D# per CDNA5 ISA ch.8 (group0 128b, group1 256b, groups2/3 zero).
    unsigned long long ga = (unsigned long long)(uintptr_t)wkvbf;   // byte address
    unsigned int lds = (unsigned int)(uintptr_t)smW;                // LDS byte offset
    tdm_u32x4 g0 = { 1u,                                            // count=1, user mode
                     lds,                                           // lds_addr [63:32]
                     (unsigned int)ga,                              // global_addr [95:64]
                     (unsigned int)((ga >> 32) & 0x01FFFFFFu) | (2u << 30) }; // addr hi + type=2
    tdm_i32x8 g1 = { (int)0x00010000,      // wg_mask=0, data_size=1 (2B)
                     (int)(256u << 16),    // tensor_dim0 = 256 (low 16 @ bit48)
                     (int)(512u << 16),    // tensor_dim0 hi=0, tensor_dim1 = 512
                     (int)(256u << 16),    // tensor_dim1 hi=0, tile_dim0 = 256
                     (int)512,             // tile_dim1 = 512, tile_dim2 = 0
                     (int)256,             // tensor_dim0_stride = 256 elems
                     0, 0 };               // stride hi, tensor_dim1_stride (unused, 2D)
    tdm_i32x4 g2 = { 0, 0, 0, 0 };
    tdm_i32x4 g3 = { 0, 0, 0, 0 };
    __builtin_amdgcn_tensor_load_to_lds(g0, g1, g2, g3, 0);
    __builtin_amdgcn_s_wait_tensorcnt(0);
  }
  __syncthreads();
#else
  {
    const u32x4* src = (const u32x4*)wkvbf;
    u32x4* dst = (u32x4*)smW;
    for (int i = tid; i < (512 * DIM) / 8; i += 256) dst[i] = src[i];
  }
  __syncthreads();
#endif

  int b = blockIdx.y;
  int t0 = blockIdx.x * 128 + wave * 16;
  const float* feat = feature + ((long)b * TLEN + t0) * DIM;
  v16bf a[8];
#pragma unroll
  for (int kk = 0; kk < 8; ++kk) a[kk] = loadA_f32cvt(feat, DIM, kk * 32);
  unsigned short* Kb = Kbuf + (long)b * TLEN * DIM;
  unsigned short* Vb = Vbuf + (long)b * DIM * TLEN;
  for (int nt = 0; nt < 32; ++nt) {
    v8f c = (v8f)0.0f;
#pragma unroll
    for (int kk = 0; kk < 8; ++kk)
      c = wmma_bf16(a[kk], loadB_u16(smW, DIM, nt * 16, kk * 32), c);
    int o0 = nt * 16;
    if (o0 < DIM) {  // K part: row-major [t][d]
#pragma unroll
      for (int j = 0; j < 8; ++j) {
        int t = t0 + half * 8 + j;  // C layout: row = VGPRj + 8*lanehalf
        Kb[(long)t * DIM + o0 + col] = f2bf(c[j]);
      }
    } else {         // V part: transposed [d][t] -> contiguous 16B store per lane
      int d = o0 - DIM + col;
      u32x4 pk;
      pk.x = (unsigned)f2bf(c[0]) | ((unsigned)f2bf(c[1]) << 16);
      pk.y = (unsigned)f2bf(c[2]) | ((unsigned)f2bf(c[3]) << 16);
      pk.z = (unsigned)f2bf(c[4]) | ((unsigned)f2bf(c[5]) << 16);
      pk.w = (unsigned)f2bf(c[6]) | ((unsigned)f2bf(c[7]) << 16);
      *(u32x4*)&Vb[(long)d * TLEN + t0 + half * 8] = pk;
    }
  }
}

// ---- Kernel 4: init output with bias ---------------------------------------
__global__ void init_out_kernel(float* out, const float* __restrict__ br) {
  if (threadIdx.x < BATCH) out[threadIdx.x] = br[0];
}

// ---- Kernel 5: fused flash-attention + residual + MLP + reduction ----------
// grid (m-tile=16, batch=16), 128 threads = 4 waves; wave w owns output cols
// [w*64, w*64+64). All waves redundantly compute the 16x32 S tile (L0-cached).
__global__ void __launch_bounds__(128)
attn_mlp_kernel(const unsigned short* __restrict__ qbf,
                const unsigned short* __restrict__ Kbuf,
                const unsigned short* __restrict__ Vbuf,
                const float* __restrict__ psub,
                const float* __restrict__ W1, const float* __restrict__ b1,
                const float* __restrict__ W2, const float* __restrict__ b2,
                const float* __restrict__ Wr, float* outp) {
  __shared__ __align__(16) float          smOA[16 * DIM];   // out_attn rows (f32)
  __shared__ __align__(16) unsigned short smH[16 * DIM];    // gelu(h1) bf16
  __shared__ __align__(16) unsigned short smP[4 * 16 * 32]; // per-wave P transpose

  int w = threadIdx.x >> 5;
  int lane = threadIdx.x & 31;
  int half = lane >> 4, col = lane & 15;
  int m0 = blockIdx.x * 16;
  int b  = blockIdx.y;
  const unsigned short* Kb = Kbuf + (long)b * TLEN * DIM;
  const unsigned short* Vb = Vbuf + (long)b * DIM * TLEN;
  unsigned short* smPw = smP + w * (16 * 32);

  // Preload Q A-tiles (16 rows x K=256 -> 8 tiles, 64 VGPRs)
  v16bf qA[8];
#pragma unroll
  for (int kk = 0; kk < 8; ++kk) qA[kk] = loadA_u16(qbf + (long)m0 * DIM, DIM, kk * 32);

  v8f o0 = (v8f)0.0f, o1 = (v8f)0.0f, o2 = (v8f)0.0f, o3 = (v8f)0.0f;
  float rm[8], ls[8];
#pragma unroll
  for (int j = 0; j < 8; ++j) { rm[j] = -__builtin_inff(); ls[j] = 0.0f; }

  const int dbase = w * 64;
  for (int t0 = 0; t0 < TLEN; t0 += 32) {
    // S = Q @ K^T (16x32), two 16x16 column tiles, K-dim = 256
    v8f s0 = (v8f)0.0f, s1 = (v8f)0.0f;
#pragma unroll
    for (int kk = 0; kk < 8; ++kk) {
      s0 = wmma_bf16(qA[kk], loadB_u16(Kb, DIM, t0,      kk * 32), s0);
      s1 = wmma_bf16(qA[kk], loadB_u16(Kb, DIM, t0 + 16, kk * 32), s1);
    }
    if (t0 + 32 < TLEN)  // pull next K tile toward L0/L2
      __builtin_prefetch(Kb + ((long)(t0 + 32) + lane) * DIM, 0, 0);

    // Online softmax. C-layout: element j of the accumulator = row (8*half + j),
    // identical mapping for S and O tiles, stats uniform across each 16-lane half.
    float alpha[8];
#pragma unroll
    for (int j = 0; j < 8; ++j) {
      float a0 = s0[j] * 0.0625f;            // 1/sqrt(256)
      float a1 = s1[j] * 0.0625f;
      float nm = fmaxf(rm[j], redmax16(fmaxf(a0, a1)));
      float al = __expf(rm[j] - nm);
      float p0 = __expf(a0 - nm);
      float p1 = __expf(a1 - nm);
      ls[j] = ls[j] * al + redsum16(p0 + p1);
      rm[j] = nm;
      alpha[j] = al;
      smPw[(half * 8 + j) * 32 + col]      = f2bf(p0);
      smPw[(half * 8 + j) * 32 + 16 + col] = f2bf(p1);
    }
#pragma unroll
    for (int j = 0; j < 8; ++j) {
      o0[j] *= alpha[j]; o1[j] *= alpha[j]; o2[j] *= alpha[j]; o3[j] *= alpha[j];
    }
    // wave-local LDS RAW fence before re-reading P in A-layout
    asm volatile("s_wait_dscnt 0" ::: "memory");
    v16bf pA = loadA_u16(smPw, 32, 0);
    // O += P @ V  (V stored transposed: contiguous 32B per lane)
    o0 = wmma_bf16(pA, loadB_u16(Vb, TLEN, dbase,      t0), o0);
    o1 = wmma_bf16(pA, loadB_u16(Vb, TLEN, dbase + 16, t0), o1);
    o2 = wmma_bf16(pA, loadB_u16(Vb, TLEN, dbase + 32, t0), o2);
    o3 = wmma_bf16(pA, loadB_u16(Vb, TLEN, dbase + 48, t0), o3);
  }

  // Finalize softmax, add pattern residual, stage out_attn rows in LDS
#pragma unroll
  for (int j = 0; j < 8; ++j) {
    float inv = 1.0f / ls[j];
    int row = half * 8 + j;
    const float* ps = psub + (long)(m0 + row) * DIM + dbase + col;
    smOA[row * DIM + dbase + col]      = o0[j] * inv + ps[0];
    smOA[row * DIM + dbase + 16 + col] = o1[j] * inv + ps[16];
    smOA[row * DIM + dbase + 32 + col] = o2[j] * inv + ps[32];
    smOA[row * DIM + dbase + 48 + col] = o3[j] * inv + ps[48];
  }
  __syncthreads();

  // MLP GEMM1: h1 = gelu(out_attn @ W1.T + b1)
  v8f c0 = (v8f)0.0f, c1 = (v8f)0.0f, c2 = (v8f)0.0f, c3 = (v8f)0.0f;
#pragma unroll
  for (int kk = 0; kk < 8; ++kk) {
    v16bf aT = loadA_f32cvt(smOA, DIM, kk * 32);
    c0 = wmma_bf16(aT, loadB_f32cvt(W1, DIM, dbase,      kk * 32), c0);
    c1 = wmma_bf16(aT, loadB_f32cvt(W1, DIM, dbase + 16, kk * 32), c1);
    c2 = wmma_bf16(aT, loadB_f32cvt(W1, DIM, dbase + 32, kk * 32), c2);
    c3 = wmma_bf16(aT, loadB_f32cvt(W1, DIM, dbase + 48, kk * 32), c3);
  }
  {
    float bb0 = b1[dbase + col], bb1 = b1[dbase + 16 + col];
    float bb2 = b1[dbase + 32 + col], bb3 = b1[dbase + 48 + col];
#pragma unroll
    for (int j = 0; j < 8; ++j) {
      int row = half * 8 + j;
      smH[row * DIM + dbase + col]      = f2bf(gelu_exact(c0[j] + bb0));
      smH[row * DIM + dbase + 16 + col] = f2bf(gelu_exact(c1[j] + bb1));
      smH[row * DIM + dbase + 32 + col] = f2bf(gelu_exact(c2[j] + bb2));
      smH[row * DIM + dbase + 48 + col] = f2bf(gelu_exact(c3[j] + bb3));
    }
  }
  __syncthreads();

  // MLP GEMM2: out_tr = h1 @ W2.T + b2 + out_attn; reduce (1/256) * out_tr . Wr
  v8f e0 = (v8f)0.0f, e1 = (v8f)0.0f, e2 = (v8f)0.0f, e3 = (v8f)0.0f;
#pragma unroll
  for (int kk = 0; kk < 8; ++kk) {
    v16bf aT = loadA_u16(smH, DIM, kk * 32);
    e0 = wmma_bf16(aT, loadB_f32cvt(W2, DIM, dbase,      kk * 32), e0);
    e1 = wmma_bf16(aT, loadB_f32cvt(W2, DIM, dbase + 16, kk * 32), e1);
    e2 = wmma_bf16(aT, loadB_f32cvt(W2, DIM, dbase + 32, kk * 32), e2);
    e3 = wmma_bf16(aT, loadB_f32cvt(W2, DIM, dbase + 48, kk * 32), e3);
  }
  float acc = 0.0f;
  {
    float bb0 = b2[dbase + col], bb1 = b2[dbase + 16 + col];
    float bb2 = b2[dbase + 32 + col], bb3 = b2[dbase + 48 + col];
    float wr0 = Wr[dbase + col], wr1 = Wr[dbase + 16 + col];
    float wr2 = Wr[dbase + 32 + col], wr3 = Wr[dbase + 48 + col];
#pragma unroll
    for (int j = 0; j < 8; ++j) {
      int row = half * 8 + j;
      acc += (e0[j] + bb0 + smOA[row * DIM + dbase + col])      * wr0;
      acc += (e1[j] + bb1 + smOA[row * DIM + dbase + 16 + col]) * wr1;
      acc += (e2[j] + bb2 + smOA[row * DIM + dbase + 32 + col]) * wr2;
      acc += (e3[j] + bb3 + smOA[row * DIM + dbase + 48 + col]) * wr3;
    }
  }
  acc = redsum32(acc);
  if (lane == 0) atomicAdd(outp + b, acc * (1.0f / 256.0f));
}

// ---- Launcher --------------------------------------------------------------
extern "C" void kernel_launch(void* const* d_in, const int* in_sizes, int n_in,
                              void* d_out, int out_size, void* d_ws, size_t ws_size,
                              hipStream_t stream) {
  const float* patterns = (const float*)d_in[0];
  const float* feature  = (const float*)d_in[1];
  const float* a_hat    = (const float*)d_in[2];
  const float* Wkv      = (const float*)d_in[3];
  const float* Wq       = (const float*)d_in[4];
  const float* W1       = (const float*)d_in[5];
  const float* b1       = (const float*)d_in[6];
  const float* W2       = (const float*)d_in[7];
  const float* b2       = (const float*)d_in[8];
  const float* Wr       = (const float*)d_in[9];
  const float* br       = (const float*)d_in[10];
  float* outp = (float*)d_out;

  // Workspace layout (needs ~134.9 MB):
  //   [0,1K)       top-k indices (256 ints)
  //   [+256K)      pattern_subset f32 [256][256]
  //   [+128K)      q bf16 [256][256]
  //   [+256K)      Wkv bf16 [512][256]   (TDM / LDS staging source)
  //   [+64M)       K bf16 [B][T][256]
  //   [+64M)       V^T bf16 [B][256][T]
  char* ws = (char*)d_ws;
  size_t off = 0;
  int* topk            = (int*)(ws + off);            off += 1024;
  float* psub          = (float*)(ws + off);          off += (size_t)256 * DIM * 4;
  unsigned short* qbf  = (unsigned short*)(ws + off); off += (size_t)256 * DIM * 2;
  unsigned short* wkvbf= (unsigned short*)(ws + off); off += (size_t)512 * DIM * 2;
  unsigned short* Kbuf = (unsigned short*)(ws + off); off += (size_t)BATCH * TLEN * DIM * 2;
  unsigned short* Vbuf = (unsigned short*)(ws + off);

  topk_kernel<<<BATCH, 256, 0, stream>>>(a_hat, topk);
  qproj_kernel<<<256, DIM, 0, stream>>>(patterns, Wq, topk, psub, qbf);
  wkvcvt_kernel<<<(512 * DIM) / 256, 256, 0, stream>>>(Wkv, wkvbf);
  kv_kernel<<<dim3(TLEN / 128, BATCH), 256, 512 * DIM * 2, stream>>>(feature, wkvbf, Kbuf, Vbuf);
  init_out_kernel<<<1, 32, 0, stream>>>(outp, br);
  attn_mlp_kernel<<<dim3(16, BATCH), 128, 0, stream>>>(qbf, Kbuf, Vbuf, psub,
                                                       W1, b1, W2, b2, Wr, outp);
}
